// MambaLayerlocal_5403068858466
// MI455X (gfx1250) — matrix operand order repaired and checked by reading, not compiled, not run on hardware
//
#include <hip/hip_runtime.h>
#include <hip/hip_bf16.h>
#include <math.h>

// Problem constants (from reference)
#define B_      2
#define NF_     8
#define H_      32
#define W_      32
#define L_      8192
#define DIM_    256
#define DINNER_ 128
#define DSTATE_ 16
#define DTRANK_ 16
#define HID_    1024
#define EPS_    1e-5f
#define BL_     (B_ * L_)   // 16384 rows for all GEMMs (divisible by 64)

typedef __bf16 bf16_t;
typedef bf16_t v16bf __attribute__((ext_vector_type(16)));
typedef bf16_t v2bf  __attribute__((ext_vector_type(2)));
typedef float  v8f   __attribute__((ext_vector_type(8)));

__device__ __forceinline__ unsigned short f2bf(float f) {
  unsigned u = __float_as_uint(f);
  u += 0x7FFFu + ((u >> 16) & 1u);   // round-to-nearest-even
  return (unsigned short)(u >> 16);
}

// Pack two f32 -> packed bf16 (one v_cvt_pk_bf16_f32 when available)
__device__ __forceinline__ unsigned cvt2bf(float a, float b) {
#if __has_builtin(__builtin_amdgcn_cvt_pk_bf16_f32)
  v2bf r = __builtin_amdgcn_cvt_pk_bf16_f32(a, b);
  return __builtin_bit_cast(unsigned, r);
#else
  return (unsigned)f2bf(a) | ((unsigned)f2bf(b) << 16);
#endif
}

// ---------------------------------------------------------------------------
// Generic WMMA GEMM: C[M,N] = epi(A[M,K] @ W[N,K]^T)
// Block tile 64(M) x 64(N), K-step 32, 8 waves (wave32). Each wave computes
// two 16x16 tiles (shared B fragment) -> 2x v_wmma_f32_16x16x32_bf16 per
// K-step. Compile-time specialization removes all bounds guards from the
// dominant call sites (KM32: K%32==0, NM64: N%64==0; M%64==0 always) and
// all epilogue pointer checks (EPI: 0=plain, 1=+bias, 2=+bias,softplus,
// 3=+bias,+residual). Fragment layouts per CDNA5 ISA 7.12.2.
// ---------------------------------------------------------------------------
template <bool KM32, bool NM64, int EPI>
__global__ __launch_bounds__(256)
void gemm_wmma(const float* __restrict__ A, int lda,
               const float* __restrict__ Wt, int ldw,
               float* __restrict__ C, int ldc,
               const float* __restrict__ bias,
               const float* __restrict__ res,
               int M, int N, int K)
{
  __shared__ __align__(16) unsigned short sA[64 * 40];  // 64 rows, pad stride 40
  __shared__ __align__(16) unsigned short sB[64 * 40];

  const int t    = threadIdx.x;
  const int m0   = blockIdx.x * 64;
  const int n0   = blockIdx.y * 64;
  const int wv   = t >> 5;
  const int lane = t & 31;
  const int nt   = wv >> 1;            // 0..3 : 16-col tile
  const int mt0  = (wv & 1) * 2;       // wave handles m-tiles mt0, mt0+1
  const int lm   = lane & 15;          // m for A-frag, n for B-frag/C
  const int hi   = lane >> 4;          // lane-half selector

  // staging assignment: 64 rows x 32 cols, 8 elems per thread
  const int row = t >> 2;              // 0..63
  const int c8  = (t & 3) * 8;         // 0,8,16,24
  const int gm  = m0 + row;            // always < M (M % 64 == 0)
  const int gn  = n0 + row;

  v8f acc0 = {}, acc1 = {};

  for (int k0 = 0; k0 < K; k0 += 32) {
    // prefetch next K tile (gfx1250 global_prefetch path)
    if (k0 + 32 < K) {
      __builtin_prefetch(A + (size_t)gm * lda + k0 + 32 + c8, 0, 3);
      if (NM64 || gn < N)
        __builtin_prefetch(Wt + (size_t)gn * ldw + k0 + 32 + c8, 0, 3);
    }

    { // ---- stage A tile (64x32 f32 -> bf16) ----
      const float* src = A + (size_t)gm * lda + k0 + c8;
      union { unsigned u[4]; uint4 q; } p;
      if constexpr (KM32) {
        float4 v0 = *(const float4*)(src);
        float4 v1 = *(const float4*)(src + 4);
        p.u[0] = cvt2bf(v0.x, v0.y); p.u[1] = cvt2bf(v0.z, v0.w);
        p.u[2] = cvt2bf(v1.x, v1.y); p.u[3] = cvt2bf(v1.z, v1.w);
      } else {
        float e[8];
        #pragma unroll
        for (int j = 0; j < 8; ++j)
          e[j] = (k0 + c8 + j < K) ? src[j] : 0.f;
        p.u[0] = cvt2bf(e[0], e[1]); p.u[1] = cvt2bf(e[2], e[3]);
        p.u[2] = cvt2bf(e[4], e[5]); p.u[3] = cvt2bf(e[6], e[7]);
      }
      *(uint4*)&sA[row * 40 + c8] = p.q;
    }
    { // ---- stage W tile (64x32 f32 -> bf16), stored [n][k] ----
      const float* src = Wt + (size_t)gn * ldw + k0 + c8;
      union { unsigned u[4]; uint4 q; } p;
      if constexpr (KM32 && NM64) {
        float4 v0 = *(const float4*)(src);
        float4 v1 = *(const float4*)(src + 4);
        p.u[0] = cvt2bf(v0.x, v0.y); p.u[1] = cvt2bf(v0.z, v0.w);
        p.u[2] = cvt2bf(v1.x, v1.y); p.u[3] = cvt2bf(v1.z, v1.w);
      } else if constexpr (KM32) {     // N-edge only (x_proj: N=48)
        if (gn < N) {
          float4 v0 = *(const float4*)(src);
          float4 v1 = *(const float4*)(src + 4);
          p.u[0] = cvt2bf(v0.x, v0.y); p.u[1] = cvt2bf(v0.z, v0.w);
          p.u[2] = cvt2bf(v1.x, v1.y); p.u[3] = cvt2bf(v1.z, v1.w);
        } else {
          p.u[0] = p.u[1] = p.u[2] = p.u[3] = 0u;
        }
      } else {                         // K-edge (dt_proj: K=16; N%64==0)
        float e[8];
        #pragma unroll
        for (int j = 0; j < 8; ++j)
          e[j] = ((NM64 || gn < N) && (k0 + c8 + j < K)) ? src[j] : 0.f;
        p.u[0] = cvt2bf(e[0], e[1]); p.u[1] = cvt2bf(e[2], e[3]);
        p.u[2] = cvt2bf(e[4], e[5]); p.u[3] = cvt2bf(e[6], e[7]);
      }
      *(uint4*)&sB[row * 40 + c8] = p.q;
    }
    __syncthreads();

    // A frag: lane(m=lm,hi): elem e -> k = (e>>3)*16 + hi*8 + (e&7)
    // B frag: lane(n=lm,hi): elem e -> k = (e>>3)*16 + hi*8 + (e&7)
    union { uint4 q[2]; v16bf v; } fb, fa0, fa1;
    fb.q[0]  = *(const uint4*)&sB[(nt * 16 + lm) * 40 + hi * 8];
    fb.q[1]  = *(const uint4*)&sB[(nt * 16 + lm) * 40 + 16 + hi * 8];
    fa0.q[0] = *(const uint4*)&sA[((mt0 + 0) * 16 + lm) * 40 + hi * 8];
    fa0.q[1] = *(const uint4*)&sA[((mt0 + 0) * 16 + lm) * 40 + 16 + hi * 8];
    fa1.q[0] = *(const uint4*)&sA[((mt0 + 1) * 16 + lm) * 40 + hi * 8];
    fa1.q[1] = *(const uint4*)&sA[((mt0 + 1) * 16 + lm) * 40 + 16 + hi * 8];

    acc0 = __builtin_amdgcn_wmma_f32_16x16x32_bf16(
        false, fa0.v, false, fb.v, (short)0, acc0, false, false);
    acc1 = __builtin_amdgcn_wmma_f32_16x16x32_bf16(
        false, fa1.v, false, fb.v, (short)0, acc1, false, false);
    __syncthreads();
  }

  // C layout: VGPR r, lane(hi,lm): m = r + 8*hi, n = lm
  const int nCol = n0 + nt * 16 + lm;
  if (NM64 || nCol < N) {
    float bv = (EPI >= 1) ? bias[nCol] : 0.f;
    auto emit = [&](const v8f& acc, int mbase) {
      #pragma unroll
      for (int r = 0; r < 8; ++r) {
        int mRow = mbase + r + 8 * hi;   // always < M
        float v = acc[r] + bv;
        if constexpr (EPI == 2) {        // branchless softplus
          float sp = log1pf(__expf(fminf(v, 20.f)));
          v = (v > 20.f) ? v : sp;
        }
        if constexpr (EPI == 3) v += res[(size_t)mRow * ldc + nCol];
        C[(size_t)mRow * ldc + nCol] = v;
      }
    };
    emit(acc0, m0 + mt0 * 16);
    emit(acc1, m0 + (mt0 + 1) * 16);
  }
}

// ---------------------------------------------------------------------------
// Hilbert gather + LayerNorm1: writes x_flat, f_flat, nx. One block per (b,i).
// ---------------------------------------------------------------------------
__global__ __launch_bounds__(256)
void prep_ln1(const float* __restrict__ x, const float* __restrict__ fx,
              const int* __restrict__ hc,
              const float* __restrict__ w, const float* __restrict__ b,
              float* __restrict__ xflat, float* __restrict__ fflat,
              float* __restrict__ nx)
{
  __shared__ float red[256];
  int bi = blockIdx.x;            // b*L + i
  int bb = bi >> 13;
  int i  = bi & (L_ - 1);
  int c  = threadIdx.x;
  int l0 = hc[i];
  int f  = l0 >> 10, hw = l0 & 1023;
  size_t src = ((size_t)(bb * NF_ + f) * DIM_ + c) * (H_ * W_) + hw;
  float xv = x[src], fv = fx[src];
  size_t dst = (size_t)bi * DIM_ + c;
  xflat[dst] = xv;
  fflat[dst] = fv;
  red[c] = xv; __syncthreads();
  for (int s = 128; s > 0; s >>= 1) { if (c < s) red[c] += red[c + s]; __syncthreads(); }
  float mean = red[0] * (1.f / DIM_); __syncthreads();
  float d = xv - mean;
  red[c] = d * d; __syncthreads();
  for (int s = 128; s > 0; s >>= 1) { if (c < s) red[c] += red[c + s]; __syncthreads(); }
  float var = red[0] * (1.f / DIM_);
  nx[dst] = d * rsqrtf(var + EPS_) * w[c] + b[c];
}

// Plain LayerNorm (LN2)
__global__ __launch_bounds__(256)
void ln_plain(const float* __restrict__ in, const float* __restrict__ w,
              const float* __restrict__ b, float* __restrict__ out)
{
  __shared__ float red[256];
  int bi = blockIdx.x;
  int c  = threadIdx.x;
  size_t idx = (size_t)bi * DIM_ + c;
  float xv = in[idx];
  red[c] = xv; __syncthreads();
  for (int s = 128; s > 0; s >>= 1) { if (c < s) red[c] += red[c + s]; __syncthreads(); }
  float mean = red[0] * (1.f / DIM_); __syncthreads();
  float d = xv - mean;
  red[c] = d * d; __syncthreads();
  for (int s = 128; s > 0; s >>= 1) { if (c < s) red[c] += red[c + s]; __syncthreads(); }
  float var = red[0] * (1.f / DIM_);
  out[idx] = d * rsqrtf(var + EPS_) * w[c] + b[c];
}

// ---------------------------------------------------------------------------
// Causal depthwise conv1d (k=4) + SiLU. dir=1 reads xz in flipped order and
// produces the sequence the reverse mamba sees (its own coordinates j).
// xi = xz[..., 0:128] (row stride 256).
// ---------------------------------------------------------------------------
__global__ void conv_silu(const float* __restrict__ xz, const float* __restrict__ cw,
                          const float* __restrict__ cb, float* __restrict__ xs, int dir)
{
  int idx = blockIdx.x * blockDim.x + threadIdx.x;  // < B*L*128 = 2^21
  int d  = idx & 127;
  int j  = (idx >> 7) & (L_ - 1);
  int bb = idx >> 20;
  float acc = cb[d];
  #pragma unroll
  for (int k = 0; k < 4; ++k) {
    int jj = j - 3 + k;
    if (jj >= 0) {
      int si = dir ? (L_ - 1 - jj) : jj;
      acc += cw[d * 4 + k] * xz[((size_t)bb * L_ + si) * DIM_ + d];
    }
  }
  xs[((size_t)bb * L_ + j) * DINNER_ + d] = acc / (1.f + __expf(-acc));
}

// ---------------------------------------------------------------------------
// Selective scan. One wave32 per 2 d-channels: lane = dloc*16 + n.
// Writes gated output in ORIGINAL sequence order (rev branch pre-flipped),
// so out_proj directly yields m_fwd / m_rev.
// ---------------------------------------------------------------------------
__global__ __launch_bounds__(32)
void scan_ssm(const float* __restrict__ dtb, const float* __restrict__ xsb,
              const float* __restrict__ xdbl, const float* __restrict__ xz,
              const float* __restrict__ Alog, const float* __restrict__ Dp,
              float* __restrict__ yg, int dir)
{
  int bb    = blockIdx.x >> 6;
  int dpair = blockIdx.x & 63;
  int lane  = threadIdx.x;
  int n     = lane & 15;
  int d     = dpair * 2 + (lane >> 4);
  float Adn = -__expf(Alog[d * DSTATE_ + n]);
  float Dd  = Dp[d];
  float h   = 0.f;
  for (int i = 0; i < L_; ++i) {
    size_t base = (size_t)bb * L_ + i;
    float dtv = dtb[base * DINNER_ + d];
    float xsv = xsb[base * DINNER_ + d];
    float Bv  = xdbl[base * 48 + DTRANK_ + n];
    float Cv  = xdbl[base * 48 + DTRANK_ + DSTATE_ + n];
    h = __expf(dtv * Adn) * h + dtv * Bv * xsv;
    float contrib = h * Cv;
    contrib += __shfl_xor(contrib, 1);   // xor masks <=8 stay inside the
    contrib += __shfl_xor(contrib, 2);   // 16-lane n-group (wave32)
    contrib += __shfl_xor(contrib, 4);
    contrib += __shfl_xor(contrib, 8);
    if (n == 0) {
      int io = dir ? (L_ - 1 - i) : i;
      size_t ob = (size_t)bb * L_ + io;
      float zv = xz[ob * DIM_ + DINNER_ + d];
      yg[ob * DINNER_ + d] = (contrib + Dd * xsv) * (zv / (1.f + __expf(-zv)));
    }
  }
}

// ---------------------------------------------------------------------------
// SSA: mean-pool over L, per-channel 3x3 mix + softmax, weighted combine.
// ---------------------------------------------------------------------------
__global__ __launch_bounds__(256)
void pool_mean(const float* __restrict__ mf, const float* __restrict__ mr,
               const float* __restrict__ ff, float* __restrict__ pooled)
{
  __shared__ float red[256];
  int c  = blockIdx.x & 255;
  int s  = (blockIdx.x >> 8) % 3;
  int bb = blockIdx.x / 768;
  const float* src = (s == 0) ? mf : (s == 1) ? mr : ff;
  float sum = 0.f;
  for (int k = 0; k < L_ / 256; ++k)
    sum += src[((size_t)bb * L_ + threadIdx.x + k * 256) * DIM_ + c];
  red[threadIdx.x] = sum; __syncthreads();
  for (int st = 128; st > 0; st >>= 1) { if (threadIdx.x < st) red[threadIdx.x] += red[threadIdx.x + st]; __syncthreads(); }
  if (threadIdx.x == 0) pooled[((size_t)bb * 3 + s) * DIM_ + c] = red[0] * (1.f / L_);
}

__global__ void ssa_weights(const float* __restrict__ pooled, const float* __restrict__ sw,
                            float* __restrict__ wout)
{
  int idx = blockIdx.x * blockDim.x + threadIdx.x;
  if (idx >= B_ * DIM_) return;
  int bb = idx >> 8, c = idx & 255;
  float wv[3];
  #pragma unroll
  for (int i = 0; i < 3; ++i) {
    float a = 0.f;
    #pragma unroll
    for (int j = 0; j < 3; ++j)
      a += sw[c * 9 + i * 3 + j] * pooled[((size_t)bb * 3 + j) * DIM_ + c];
    wv[i] = a;
  }
  float mx = fmaxf(wv[0], fmaxf(wv[1], wv[2]));
  float e0 = __expf(wv[0] - mx), e1 = __expf(wv[1] - mx), e2 = __expf(wv[2] - mx);
  float inv = 1.f / (e0 + e1 + e2);
  wout[((size_t)bb * 3 + 0) * DIM_ + c] = e0 * inv;
  wout[((size_t)bb * 3 + 1) * DIM_ + c] = e1 * inv;
  wout[((size_t)bb * 3 + 2) * DIM_ + c] = e2 * inv;
}

__global__ void ssa_combine(const float* __restrict__ xf, const float* __restrict__ ff,
                            const float* __restrict__ mf, const float* __restrict__ mr,
                            const float* __restrict__ w, float* __restrict__ xm)
{
  size_t idx = (size_t)blockIdx.x * blockDim.x + threadIdx.x;  // < B*L*256 = 2^22
  int c  = idx & 255;
  int bb = (int)(idx >> 21);
  float w0 = w[((size_t)bb * 3 + 0) * DIM_ + c];
  float w1 = w[((size_t)bb * 3 + 1) * DIM_ + c];
  float w2 = w[((size_t)bb * 3 + 2) * DIM_ + c];
  xm[idx] = xf[idx] + w0 * mf[idx] + w1 * mr[idx] + w2 * ff[idx];
}

// ---------------------------------------------------------------------------
// Depthwise 3x3x3 conv over (nf,h,w) + exact GELU. hdn stored (b,L,HID).
// ---------------------------------------------------------------------------
__global__ void dwconv_gelu(const float* __restrict__ hdn, const float* __restrict__ dww,
                            const float* __restrict__ dwb, float* __restrict__ hact)
{
  size_t idx = (size_t)blockIdx.x * blockDim.x + threadIdx.x;  // < B*L*HID = 2^24
  int c  = idx & 1023;
  int l  = (int)(idx >> 10) & (L_ - 1);
  int bb = (int)(idx >> 23);
  int f = l >> 10, hh = (l >> 5) & 31, ww = l & 31;
  float acc = dwb[c];
  #pragma unroll
  for (int kd = -1; kd <= 1; ++kd) {
    int fz = f + kd; if (fz < 0 || fz >= NF_) continue;
    #pragma unroll
    for (int kh = -1; kh <= 1; ++kh) {
      int hz = hh + kh; if (hz < 0 || hz >= H_) continue;
      #pragma unroll
      for (int kw = -1; kw <= 1; ++kw) {
        int wz = ww + kw; if (wz < 0 || wz >= W_) continue;
        int lp = (fz << 10) | (hz << 5) | wz;
        acc += dww[c * 27 + (kd + 1) * 9 + (kh + 1) * 3 + (kw + 1)]
             * hdn[((size_t)bb * L_ + lp) * HID_ + c];
      }
    }
  }
  hact[idx] = 0.5f * acc * (1.f + erff(acc * 0.70710678118654752f));
}

// Inverse hilbert scatter into output (b, nf, c, h, w)
__global__ void scatter_out(const float* __restrict__ xo, const int* __restrict__ hc,
                            float* __restrict__ out)
{
  size_t idx = (size_t)blockIdx.x * blockDim.x + threadIdx.x;  // < 2^22
  int c  = idx & 255;
  int i  = (int)(idx >> 8) & (L_ - 1);
  int bb = (int)(idx >> 21);
  int l0 = hc[i];
  int f = l0 >> 10, hw = l0 & 1023;
  out[((size_t)(bb * NF_ + f) * DIM_ + c) * (H_ * W_) + hw] = xo[idx];
}

// ---------------------------------------------------------------------------
// Workspace layout (floats). Total ~53.2M floats ~= 203 MB.
//   hact aliases [0, 16M) (x_flat/f_flat/nx/xz all dead by then);
//   xo aliases m_fwd (dead after ssa_combine); nxm aliases nx.
// ---------------------------------------------------------------------------
static const size_t A_xflat = 0;
static const size_t A_fflat = 4194304;
static const size_t A_nx    = 8388608;    // also nxm
static const size_t A_xz    = 12582912;
static const size_t A_hact  = 0;          // alias, 16777216 floats
static const size_t A_xs    = 16777216;   // 2097152
static const size_t A_xdbl  = 18874368;   // 786432
static const size_t A_dt    = 19660800;   // 2097152
static const size_t A_yg    = 21757952;   // 2097152
static const size_t A_mf    = 23855104;   // 4194304 (also xo)
static const size_t A_mr    = 28049408;   // 4194304
static const size_t A_pool  = 32243712;   // 1536
static const size_t A_wgt   = 32245248;   // 1536
static const size_t A_xm    = 32246784;   // 4194304
static const size_t A_hdn   = 36441088;   // 16777216

extern "C" void kernel_launch(void* const* d_in, const int* in_sizes, int n_in,
                              void* d_out, int out_size, void* d_ws, size_t ws_size,
                              hipStream_t stream)
{
  const float* x     = (const float*)d_in[0];
  const float* fx    = (const float*)d_in[1];
  const int*   hc    = (const int*)  d_in[2];
  const float* ln1w  = (const float*)d_in[3];
  const float* ln1b  = (const float*)d_in[4];
  const float* ln2w  = (const float*)d_in[5];
  const float* ln2b  = (const float*)d_in[6];
  const float* inpw  = (const float*)d_in[7];   // (256, 256)
  const float* convw = (const float*)d_in[8];   // (128, 1, 4)
  const float* convb = (const float*)d_in[9];
  const float* xpw   = (const float*)d_in[10];  // (48, 128)
  const float* dtpw  = (const float*)d_in[11];  // (128, 16)
  const float* dtpb  = (const float*)d_in[12];
  const float* Alog  = (const float*)d_in[13];  // (128, 16)
  const float* Dp    = (const float*)d_in[14];
  const float* outpw = (const float*)d_in[15];  // (256, 128)
  const float* ssaw  = (const float*)d_in[16];  // (256, 3, 3)
  const float* fc1w  = (const float*)d_in[17];  // (1024, 256)
  const float* fc1b  = (const float*)d_in[18];
  const float* dww   = (const float*)d_in[19];  // (1024, 1, 3, 3, 3)
  const float* dwb   = (const float*)d_in[20];
  const float* fc2w  = (const float*)d_in[21];  // (256, 1024)
  const float* fc2b  = (const float*)d_in[22];

  float* ws     = (float*)d_ws;
  float* xflat  = ws + A_xflat;
  float* fflat  = ws + A_fflat;
  float* nx     = ws + A_nx;
  float* xz     = ws + A_xz;
  float* xs     = ws + A_xs;
  float* xdbl   = ws + A_xdbl;
  float* dtb    = ws + A_dt;
  float* yg     = ws + A_yg;
  float* mf     = ws + A_mf;
  float* mr     = ws + A_mr;
  float* pooled = ws + A_pool;
  float* wgt    = ws + A_wgt;
  float* xm     = ws + A_xm;
  float* hdn    = ws + A_hdn;
  float* hact   = ws + A_hact;
  float* nxm    = ws + A_nx;   // alias (nx dead after in_proj)
  float* xo     = ws + A_mf;   // alias (m_fwd dead after ssa_combine)

  // 1) gather + LN1
  prep_ln1<<<BL_, 256, 0, stream>>>(x, fx, hc, ln1w, ln1b, xflat, fflat, nx);

  // 2) in_proj (shared by fwd & rev branches): (16384,256)x(256,256)^T
  gemm_wmma<true, true, 0><<<dim3(BL_ / 64, DIM_ / 64), 256, 0, stream>>>(
      nx, DIM_, inpw, DIM_, xz, DIM_, nullptr, nullptr, BL_, DIM_, DIM_);

  // 3) per-direction: conv -> x_proj -> dt_proj -> scan -> out_proj
  for (int dir = 0; dir < 2; ++dir) {
    conv_silu<<<(B_ * L_ * DINNER_) / 256, 256, 0, stream>>>(xz, convw, convb, xs, dir);
    gemm_wmma<true, false, 0><<<dim3(BL_ / 64, 1), 256, 0, stream>>>(
        xs, DINNER_, xpw, DINNER_, xdbl, 48, nullptr, nullptr, BL_, 48, DINNER_);
    gemm_wmma<false, true, 2><<<dim3(BL_ / 64, DINNER_ / 64), 256, 0, stream>>>(
        xdbl, 48, dtpw, DTRANK_, dtb, DINNER_, dtpb, nullptr, BL_, DINNER_, DTRANK_);
    scan_ssm<<<B_ * DINNER_ / 2, 32, 0, stream>>>(dtb, xs, xdbl, xz, Alog, Dp, yg, dir);
    gemm_wmma<true, true, 0><<<dim3(BL_ / 64, DIM_ / 64), 256, 0, stream>>>(
        yg, DINNER_, outpw, DINNER_, dir ? mr : mf, DIM_, nullptr, nullptr,
        BL_, DIM_, DINNER_);
  }

  // 4) SSA mix + residual
  pool_mean<<<B_ * 3 * DIM_, 256, 0, stream>>>(mf, mr, fflat, pooled);
  ssa_weights<<<(B_ * DIM_ + 255) / 256, 256, 0, stream>>>(pooled, ssaw, wgt);
  ssa_combine<<<(B_ * L_ * DIM_) / 256, 256, 0, stream>>>(xflat, fflat, mf, mr, wgt, xm);

  // 5) MLP: LN2 -> fc1 -> dwconv3d+gelu -> fc2 (+bias, +residual xm)
  ln_plain<<<BL_, 256, 0, stream>>>(xm, ln2w, ln2b, nxm);
  gemm_wmma<true, true, 1><<<dim3(BL_ / 64, HID_ / 64), 256, 0, stream>>>(
      nxm, DIM_, fc1w, DIM_, hdn, HID_, fc1b, nullptr, BL_, HID_, DIM_);
  dwconv_gelu<<<(B_ * L_ * HID_) / 256, 256, 0, stream>>>(hdn, dww, dwb, hact);
  gemm_wmma<true, true, 3><<<dim3(BL_ / 64, DIM_ / 64), 256, 0, stream>>>(
      hact, HID_, fc2w, HID_, xo, DIM_, fc2b, xm, BL_, DIM_, HID_);

  // 6) inverse hilbert scatter to output
  scatter_out<<<(B_ * L_ * DIM_) / 256, 256, 0, stream>>>(xo, hc, (float*)d_out);
}
